// GnnPF_13082470383785
// MI455X (gfx1250) — compile-verified, matrix operand order checked by reading, not run on hardware
//
#include <hip/hip_runtime.h>
#include <hip/hip_bf16.h>

typedef __attribute__((ext_vector_type(16))) __bf16 v16bf;
typedef __attribute__((ext_vector_type(8)))  __bf16 v8bf;
typedef __attribute__((ext_vector_type(8)))  float  v8f;

#define CDIV(a,b) (((a)+(b)-1)/(b))

__device__ inline unsigned short f2bf(float f){
  unsigned int u = __float_as_uint(f);
  u += 0x7FFFu + ((u >> 16) & 1u);            // round-to-nearest-even
  return (unsigned short)(u >> 16);
}
__device__ inline float lrelu(float v){ return v > 0.f ? v : 0.2f * v; }

// ---------------- elementwise / conversion kernels ----------------
__global__ void k_zero_f(float* p, int n){
  int i = blockIdx.x*blockDim.x + threadIdx.x; if (i < n) p[i] = 0.f;
}
__global__ void k_zero_i(int* p, int n){
  int i = blockIdx.x*blockDim.x + threadIdx.x; if (i < n) p[i] = 0;
}
__global__ void k_copy_i(const int* __restrict__ a, int* __restrict__ b, int n){
  int i = blockIdx.x*blockDim.x + threadIdx.x; if (i < n) b[i] = a[i];
}
__global__ void k_set_u8(unsigned char* p, int v, int n){
  int i = blockIdx.x*blockDim.x + threadIdx.x; if (i < n) p[i] = (unsigned char)v;
}
__global__ void k_cvt_bf16(const float* __restrict__ in, unsigned short* __restrict__ out, int n){
  int i = blockIdx.x*blockDim.x + threadIdx.x; if (i < n) out[i] = f2bf(in[i]);
}
// out[N][K] = (bf16) in[K][N]
__global__ void k_transpose_cvt(const float* __restrict__ in, unsigned short* __restrict__ out,
                                int K, int N){
  int i = blockIdx.x*blockDim.x + threadIdx.x;
  if (i >= N*K) return;
  int nn = i / K, k = i - nn*K;
  out[i] = f2bf(in[(size_t)k*N + nn]);
}
__global__ void k_build_aseq(const float* __restrict__ seq, unsigned short* __restrict__ out, int Nn){
  int i = blockIdx.x*blockDim.x + threadIdx.x;
  if (i >= Nn*32) return;
  int l = i >> 5, c = i & 31;
  out[i] = (c < 25) ? f2bf(seq[(size_t)c*Nn + l]) : (unsigned short)0;
}
__global__ void k_build_seqwt(const float* __restrict__ w, unsigned short* __restrict__ out){
  int i = blockIdx.x*blockDim.x + threadIdx.x;
  if (i >= 512*32) return;
  int o = i >> 5, c = i & 31;
  out[i] = (c < 25) ? f2bf(w[o*25 + c]) : (unsigned short)0;
}

// ---------------- CSR-by-dst build ----------------
__global__ void k_deg(const int* __restrict__ dst, int* __restrict__ deg, int E){
  int e = blockIdx.x*blockDim.x + threadIdx.x; if (e < E) atomicAdd(&deg[dst[e]], 1);
}
// single block of 256 threads; Nn multiple of 256
__global__ void k_scan(const int* __restrict__ deg, int* __restrict__ rowptr, int Nn){
  __shared__ int pre[257];
  __shared__ int tot[256];
  int t = threadIdx.x;
  int chunk = Nn / 256;
  int s = 0;
  for (int i = 0; i < chunk; i++) s += deg[t*chunk + i];
  tot[t] = s; __syncthreads();
  if (t == 0){ pre[0] = 0; for (int i = 0; i < 256; i++) pre[i+1] = pre[i] + tot[i]; }
  __syncthreads();
  int run = pre[t];
  for (int i = 0; i < chunk; i++){ rowptr[t*chunk + i] = run; run += deg[t*chunk + i]; }
  if (t == 255) rowptr[Nn] = run;
}
__global__ void k_fill(const int* __restrict__ dst, int* __restrict__ cursor,
                       int* __restrict__ eidx, int E){
  int e = blockIdx.x*blockDim.x + threadIdx.x; if (e >= E) return;
  int pos = atomicAdd(&cursor[dst[e]], 1);
  eidx[pos] = e;
}

// ---------------- WMMA GEMM (blocked): C[M][N] = A[M][K]*Bt[N][K]^T ----------------
// wave computes a 32x64 macro-tile = 2x4 WMMA tiles (64 acc VGPRs -> no spills);
// M%32==0, N%64==0, K%32==0
union FragU { v16bf v; v8bf h[2]; };
__global__ void k_wmma_gemm_b(const __bf16* __restrict__ A, const __bf16* __restrict__ Bt,
                              float* __restrict__ C, const float* __restrict__ bias,
                              int M, int N, int K, int relu){
  int tiles_n = N >> 6, tiles_m = M >> 5;
  int tile = blockIdx.x * (blockDim.x >> 5) + (threadIdx.x >> 5);   // wave-uniform
  if (tile >= tiles_m * tiles_n) return;
  int tm = tile / tiles_n, tn = tile - tm * tiles_n;
  int lane = threadIdx.x & 31;
  int hl = lane >> 4;
  int r  = lane & 15;
  const __bf16* arow[2]; const __bf16* brow[4];
  #pragma unroll
  for (int i = 0; i < 2; i++)
    arow[i] = A  + (size_t)(tm*32 + i*16 + r) * K + hl*8;
  #pragma unroll
  for (int j = 0; j < 4; j++)
    brow[j] = Bt + (size_t)(tn*64 + j*16 + r) * K + hl*8;
  v8f zero = {};
  v8f acc[2][4];
  #pragma unroll
  for (int i = 0; i < 2; i++)
    #pragma unroll
    for (int j = 0; j < 4; j++) acc[i][j] = zero;
  for (int k0 = 0; k0 < K; k0 += 32){
    FragU ua[2], ub[4];
    #pragma unroll
    for (int i = 0; i < 2; i++){
      __builtin_prefetch(arow[i] + k0 + 256, 0, 1);   // global_prefetch_b8
      ua[i].h[0] = *(const v8bf*)(arow[i] + k0);
      ua[i].h[1] = *(const v8bf*)(arow[i] + k0 + 16);
    }
    #pragma unroll
    for (int j = 0; j < 4; j++){
      __builtin_prefetch(brow[j] + k0 + 256, 0, 1);
      ub[j].h[0] = *(const v8bf*)(brow[j] + k0);
      ub[j].h[1] = *(const v8bf*)(brow[j] + k0 + 16);
    }
    #pragma unroll
    for (int i = 0; i < 2; i++)
      #pragma unroll
      for (int j = 0; j < 4; j++)
        acc[i][j] = __builtin_amdgcn_wmma_f32_16x16x32_bf16(
            false, ua[i].v, false, ub[j].v, (short)0, acc[i][j], false, false);
  }
  #pragma unroll
  for (int i = 0; i < 2; i++){
    int cm0 = tm*32 + i*16 + hl*8;
    #pragma unroll
    for (int j = 0; j < 4; j++){
      int cn = tn*64 + j*16 + r;
      float bv = bias ? bias[cn] : 0.f;
      #pragma unroll
      for (int q = 0; q < 8; q++){
        float v = acc[i][j][q] + bv;
        if (relu) v = v > 0.f ? v : 0.f;
        C[(size_t)(cm0 + q) * N + cn] = v;
      }
    }
  }
}

// simple 1-tile-per-wave variant for small M (classifier GEMMs, M=16)
__global__ void k_wmma_gemm_s(const __bf16* __restrict__ A, const __bf16* __restrict__ Bt,
                              float* __restrict__ C, const float* __restrict__ bias,
                              int M, int N, int K, int relu){
  int tiles_n = N >> 4, tiles_m = M >> 4;
  int tile = blockIdx.x * (blockDim.x >> 5) + (threadIdx.x >> 5);
  if (tile >= tiles_m * tiles_n) return;
  int tm = tile / tiles_n, tn = tile - tm * tiles_n;
  int lane = threadIdx.x & 31;
  int hl = lane >> 4, r = lane & 15;
  const __bf16* arow = A  + (size_t)(tm*16 + r) * K + hl*8;
  const __bf16* brow = Bt + (size_t)(tn*16 + r) * K + hl*8;
  v8f acc = {};
  for (int k0 = 0; k0 < K; k0 += 32){
    __builtin_prefetch(arow + k0 + 128, 0, 1);
    __builtin_prefetch(brow + k0 + 128, 0, 1);
    FragU ua, ub;
    ua.h[0] = *(const v8bf*)(arow + k0);
    ua.h[1] = *(const v8bf*)(arow + k0 + 16);
    ub.h[0] = *(const v8bf*)(brow + k0);
    ub.h[1] = *(const v8bf*)(brow + k0 + 16);
    acc = __builtin_amdgcn_wmma_f32_16x16x32_bf16(false, ua.v, false, ub.v,
                                                  (short)0, acc, false, false);
  }
  int cn = tn*16 + r, cm0 = tm*16 + hl*8;
  float bv = bias ? bias[cn] : 0.f;
  #pragma unroll
  for (int i = 0; i < 8; i++){
    float v = acc[i] + bv;
    if (relu) v = v > 0.f ? v : 0.f;
    C[(size_t)(cm0 + i) * N + cn] = v;
  }
}

// ---------------- GAT attention ----------------
// wave per node: two dot products against per-head attention vectors
__global__ void k_rowdot2(const float* __restrict__ H, const float* __restrict__ a1,
                          const float* __restrict__ a2, float* __restrict__ d1,
                          float* __restrict__ d2, int Nn, int Co){
  int nv = blockIdx.x * (blockDim.x >> 5) + (threadIdx.x >> 5);
  if (nv >= Nn) return;
  int lane = threadIdx.x & 31;
  const float* row = H + (size_t)nv * Co;
  float s1 = 0.f, s2 = 0.f;
  for (int o = lane; o < Co; o += 32){ float hv = row[o]; s1 += hv*a1[o]; s2 += hv*a2[o]; }
  for (int off = 16; off; off >>= 1){ s1 += __shfl_down(s1, off, 32); s2 += __shfl_down(s2, off, 32); }
  if (lane == 0){ d1[nv] = s1; d2[nv] = s2; }
}

// fused per-head GAT aggregation: wave per destination node, CSR gather, no atomics.
// xn[n,:] += mean-head contribution (num/den)/12
template<int CO>
__global__ void k_gat_gather(const int* __restrict__ rowptr, const int* __restrict__ eidx,
                             const int* __restrict__ src, const unsigned char* __restrict__ emask,
                             const float* __restrict__ als, const float* __restrict__ ald,
                             const float* __restrict__ H, float* __restrict__ xn, int Nn){
  int n = blockIdx.x * (blockDim.x >> 5) + (threadIdx.x >> 5);
  if (n >= Nn) return;
  int lane = threadIdx.x & 31;
  int beg = rowptr[n], end = rowptr[n+1];
  float aldn = ald[n];
  float eself = lrelu(als[n] + aldn);
  // pass 1: running max (lane-strided, then xor-reduce so all lanes hold m)
  float m = eself;
  for (int q = beg + lane; q < end; q += 32){
    int e = eidx[q]; int s = src[e];
    if (emask[e] && s != n) m = fmaxf(m, lrelu(als[s] + aldn));
  }
  #pragma unroll
  for (int off = 16; off; off >>= 1) m = fmaxf(m, __shfl_xor(m, off, 32));
  // pass 2: softmax weights + weighted sum of source rows (edges scalar per wave)
  float pself = expf(eself - m);
  float den = pself;
  constexpr int CL = CO / 32;
  float acc[CL];
  #pragma unroll
  for (int i = 0; i < CL; i++) acc[i] = 0.f;
  for (int q = beg; q < end; q++){
    int e = eidx[q]; int s = src[e];
    if (!emask[e] || s == n) continue;             // wave-uniform
    float p = expf(lrelu(als[s] + aldn) - m);
    den += p;                                      // identical on every lane
    const float* hs = H + (size_t)s * CO;
    #pragma unroll
    for (int i = 0; i < CL; i++) acc[i] += p * hs[lane + i*32];
  }
  const float* hn = H + (size_t)n * CO;
  float* xr = xn + (size_t)n * CO;
  float inv = 1.f / (den * 12.f);
  #pragma unroll
  for (int i = 0; i < CL; i++)
    xr[lane + i*32] += (acc[i] + pself * hn[lane + i*32]) * inv;
}

// fused SAG score: msg gather + lin_rel + lin_root, wave per node, no atomics
template<int CO>
__global__ void k_sag_gather(const int* __restrict__ rowptr, const int* __restrict__ eidx,
                             const int* __restrict__ src, const unsigned char* __restrict__ emask,
                             const float* __restrict__ x, const float* __restrict__ wrel,
                             const float* __restrict__ brel, const float* __restrict__ wroot,
                             float* __restrict__ raw, int Nn){
  int n = blockIdx.x * (blockDim.x >> 5) + (threadIdx.x >> 5);
  if (n >= Nn) return;
  int lane = threadIdx.x & 31;
  int beg = rowptr[n], end = rowptr[n+1];
  constexpr int CL = CO / 32;
  float acc[CL];
  #pragma unroll
  for (int i = 0; i < CL; i++) acc[i] = 0.f;
  for (int q = beg; q < end; q++){
    int e = eidx[q];
    if (!emask[e]) continue;                       // wave-uniform (self-loops included)
    const float* xs = x + (size_t)src[e] * CO;
    #pragma unroll
    for (int i = 0; i < CL; i++) acc[i] += xs[lane + i*32];
  }
  const float* xr = x + (size_t)n * CO;
  float s = 0.f;
  #pragma unroll
  for (int i = 0; i < CL; i++){
    int o = lane + i*32;
    s += acc[i]*wrel[o] + xr[o]*wroot[o];
  }
  #pragma unroll
  for (int off = 16; off; off >>= 1) s += __shfl_xor(s, off, 32);
  if (lane == 0) raw[n] = s + brel[0];
}

// ---------------- SAG top-k pooling ----------------
__global__ void k_sag_pool(const float* __restrict__ raw, const unsigned char* __restrict__ nmask,
                           unsigned char* __restrict__ newmask, int kkeep){
  __shared__ float s[512];
  __shared__ short idx[512];
  int g = blockIdx.x, t = threadIdx.x;
  for (int e = t; e < 512; e += blockDim.x){
    int n = g*512 + e;
    s[e] = nmask[n] ? tanhf(raw[n]) : -3.0e38f;
    idx[e] = (short)e;
  }
  __syncthreads();
  for (int ksz = 2; ksz <= 512; ksz <<= 1){
    for (int j = ksz >> 1; j > 0; j >>= 1){
      for (int e = t; e < 512; e += blockDim.x){
        int p = e ^ j;
        if (p > e){
          bool up = ((e & ksz) == 0);
          bool sw = up ? (s[e] < s[p]) : (s[e] > s[p]);   // descending
          if (sw){ float tv=s[e]; s[e]=s[p]; s[p]=tv; short ti=idx[e]; idx[e]=idx[p]; idx[p]=ti; }
        }
      }
      __syncthreads();
    }
  }
  for (int e = t; e < 512; e += blockDim.x)
    if (e < kkeep) newmask[g*512 + idx[e]] = 1;
}
__global__ void k_pool_apply(float* __restrict__ x, const float* __restrict__ raw,
                             const unsigned char* __restrict__ newmask, int Nn, int Co){
  int i = blockIdx.x*blockDim.x + threadIdx.x; if (i >= Nn*Co) return;
  int n = i / Co;
  x[i] = newmask[n] ? x[i] * tanhf(raw[n]) : 0.f;
}
__global__ void k_edge_mask_update(const int* __restrict__ src, const int* __restrict__ dst,
                                   const unsigned char* __restrict__ nm,
                                   unsigned char* __restrict__ emask, int E){
  int e = blockIdx.x*blockDim.x + threadIdx.x; if (e >= E) return;
  emask[e] = (unsigned char)(emask[e] && nm[src[e]] && nm[dst[e]]);
}

// ---------------- readout ----------------
__global__ void k_pooled(const float* __restrict__ x, float* __restrict__ feat, int Co, int NPER){
  int i = blockIdx.x*blockDim.x + threadIdx.x;
  if (i >= 16*Co) return;
  int b = i / Co, o = i - b*Co;
  float s = 0.f;
  for (int j = 0; j < NPER; j++) s += x[((size_t)(b*NPER + j))*Co + o];
  feat[(size_t)b*2304 + o] = s * (1.f / 32.f);   // n_per >> 4
}
__global__ void k_copy_se(const float* __restrict__ se, float* __restrict__ feat, int Bg){
  int i = blockIdx.x*blockDim.x + threadIdx.x;
  if (i >= Bg*1280) return;
  int b = i / 1280, j = i - b*1280;
  feat[(size_t)b*2304 + 1024 + j] = se[i];
}

// ---------------- host orchestration ----------------
extern "C" void kernel_launch(void* const* d_in, const int* in_sizes, int n_in,
                              void* d_out, int out_size, void* d_ws, size_t ws_size,
                              hipStream_t stream){
  (void)in_sizes; (void)n_in; (void)out_size; (void)ws_size;
  const int Nn = 8192, Eg = 131072, Bg = 16, HEADS = 12;
  const int ci_[4] = {512,512,512,1024}, co_[4] = {512,512,1024,1024};
  const int kk_[4] = {256,128,64,32};

  // inputs (setup_inputs order; params flattened as a jax pytree: dict keys sorted)
  const float* seq    = (const float*)d_in[1];
  const int*   src    = (const int*)d_in[3];
  const int*   dst    = src + Eg;
  const float* seqemb = (const float*)d_in[4];
  const float* cls_b1 = (const float*)d_in[6];
  const float* cls_b2 = (const float*)d_in[7];
  const float* cls_w1 = (const float*)d_in[8];
  const float* cls_w2 = (const float*)d_in[9];
  const float *gW[4], *gAd[4], *gAs[4];
  { int pi = 12;                                   // gat[l]: {W, a_dst, a_src}
    for (int l = 0; l < 4; l++){ gW[l]=(const float*)d_in[pi]; gAd[l]=(const float*)d_in[pi+1];
                                 gAs[l]=(const float*)d_in[pi+2]; pi += 3; } }
  const float *sB[4], *sWrel[4], *sWroot[4];
  { int pi = 26;                                   // sag[l]: {b_rel, w_rel, w_root}
    for (int l = 0; l < 4; l++){ sB[l]=(const float*)d_in[pi]; sWrel[l]=(const float*)d_in[pi+1];
                                 sWroot[l]=(const float*)d_in[pi+2]; pi += 3; } }
  const float* seq_b = (const float*)d_in[38];
  const float* seq_w = (const float*)d_in[39];

  // workspace carve (256B aligned)
  char* wp = (char*)d_ws;
  auto carve = [&](size_t bytes)->void*{ void* p=(void*)wp; wp += (bytes + 255) & ~(size_t)255; return p; };
  float* x0   = (float*)carve((size_t)Nn*1024*4);
  float* x1   = (float*)carve((size_t)Nn*1024*4);
  unsigned short* xb = (unsigned short*)carve((size_t)Nn*1024*2);
  float* hH   = (float*)carve((size_t)Nn*1024*4);
  unsigned short* wt = (unsigned short*)carve((size_t)2752*2752*2);
  float* als  = (float*)carve((size_t)Nn*4);
  float* ald  = (float*)carve((size_t)Nn*4);
  float* raw  = (float*)carve((size_t)Nn*4);
  int*   deg    = (int*)carve((size_t)Nn*4);
  int*   rowptr = (int*)carve((size_t)(Nn+1)*4);
  int*   cursor = (int*)carve((size_t)Nn*4);
  int*   eidx   = (int*)carve((size_t)Eg*4);
  unsigned char* emask = (unsigned char*)carve(Eg);
  unsigned char* nmA   = (unsigned char*)carve(Nn);
  unsigned char* nmB   = (unsigned char*)carve(Nn);
  float* feat = (float*)carve((size_t)Bg*2304*4);
  unsigned short* featb = (unsigned short*)carve((size_t)Bg*2304*2);
  float* hdn  = (float*)carve((size_t)Bg*2752*4);
  unsigned short* hdnb  = (unsigned short*)carve((size_t)Bg*2752*2);

  dim3 blk(256);
  // masks + CSR-by-dst (edge list is fixed; masks applied at gather time)
  k_set_u8<<<CDIV(Eg,256), blk, 0, stream>>>(emask, 1, Eg);
  k_set_u8<<<CDIV(Nn,256), blk, 0, stream>>>(nmA, 1, Nn);
  k_zero_i<<<CDIV(Nn,256), blk, 0, stream>>>(deg, Nn);
  k_deg<<<CDIV(Eg,256), blk, 0, stream>>>(dst, deg, Eg);
  k_scan<<<1, blk, 0, stream>>>(deg, rowptr, Nn);
  k_copy_i<<<CDIV(Nn,256), blk, 0, stream>>>(rowptr, cursor, Nn);
  k_fill<<<CDIV(Eg,256), blk, 0, stream>>>(dst, cursor, eidx, Eg);

  // x0 = relu(seq^T @ seq_w^T + seq_b)   [8192 x 512], K padded 25->32
  k_build_aseq<<<CDIV(Nn*32,256), blk, 0, stream>>>(seq, xb, Nn);
  k_build_seqwt<<<CDIV(512*32,256), blk, 0, stream>>>(seq_w, wt);
  k_wmma_gemm_b<<<CDIV((Nn/32)*(512/64),8), blk, 0, stream>>>(
      (const __bf16*)xb, (const __bf16*)wt, x0, seq_b, Nn, 512, 32, 1);

  unsigned char* nmask = nmA; unsigned char* nmNew = nmB;
  float* xc = x0; float* xn = x1;
  for (int l = 0; l < 4; l++){
    int ci = ci_[l], co = co_[l];
    k_cvt_bf16<<<CDIV(Nn*ci,256), blk, 0, stream>>>(xc, xb, Nn*ci);
    k_zero_f<<<CDIV(Nn*co,256), blk, 0, stream>>>(xn, Nn*co);
    for (int h = 0; h < HEADS; h++){
      k_transpose_cvt<<<CDIV(ci*co,256), blk, 0, stream>>>(gW[l] + (size_t)h*ci*co, wt, ci, co);
      k_wmma_gemm_b<<<CDIV((Nn/32)*(co/64),8), blk, 0, stream>>>(
          (const __bf16*)xb, (const __bf16*)wt, hH, nullptr, Nn, co, ci, 0);
      k_rowdot2<<<CDIV(Nn,8), blk, 0, stream>>>(hH, gAs[l] + (size_t)h*co, gAd[l] + (size_t)h*co,
                                                als, ald, Nn, co);
      if (co == 512)
        k_gat_gather<512><<<CDIV(Nn,8), blk, 0, stream>>>(rowptr, eidx, src, emask, als, ald, hH, xn, Nn);
      else
        k_gat_gather<1024><<<CDIV(Nn,8), blk, 0, stream>>>(rowptr, eidx, src, emask, als, ald, hH, xn, Nn);
    }
    { float* t = xc; xc = xn; xn = t; }
    // SAG score (fused gather) + top-k pool
    if (co == 512)
      k_sag_gather<512><<<CDIV(Nn,8), blk, 0, stream>>>(rowptr, eidx, src, emask, xc,
                                                        sWrel[l], sB[l], sWroot[l], raw, Nn);
    else
      k_sag_gather<1024><<<CDIV(Nn,8), blk, 0, stream>>>(rowptr, eidx, src, emask, xc,
                                                         sWrel[l], sB[l], sWroot[l], raw, Nn);
    k_set_u8<<<CDIV(Nn,256), blk, 0, stream>>>(nmNew, 0, Nn);
    k_sag_pool<<<Bg, blk, 0, stream>>>(raw, nmask, nmNew, kk_[l]);
    k_pool_apply<<<CDIV(Nn*co,256), blk, 0, stream>>>(xc, raw, nmNew, Nn, co);
    k_edge_mask_update<<<CDIV(Eg,256), blk, 0, stream>>>(src, dst, nmNew, emask, Eg);
    { unsigned char* t = nmask; nmask = nmNew; nmNew = t; }
  }

  // readout: mean-pool, concat seq_embed, 2-layer MLP (WMMA, M=16 -> simple kernel)
  k_pooled<<<CDIV(Bg*1024,256), blk, 0, stream>>>(xc, feat, 1024, 512);
  k_copy_se<<<CDIV(Bg*1280,256), blk, 0, stream>>>(seqemb, feat, Bg);
  k_cvt_bf16<<<CDIV(Bg*2304,256), blk, 0, stream>>>(feat, featb, Bg*2304);
  k_transpose_cvt<<<CDIV(2304*2752,256), blk, 0, stream>>>(cls_w1, wt, 2304, 2752);
  k_wmma_gemm_s<<<CDIV((Bg/16)*(2752/16),8), blk, 0, stream>>>(
      (const __bf16*)featb, (const __bf16*)wt, hdn, cls_b1, Bg, 2752, 2304, 1);
  k_cvt_bf16<<<CDIV(Bg*2752,256), blk, 0, stream>>>(hdn, hdnb, Bg*2752);
  k_transpose_cvt<<<CDIV(2752*2752,256), blk, 0, stream>>>(cls_w2, wt, 2752, 2752);
  k_wmma_gemm_s<<<CDIV((Bg/16)*(2752/16),8), blk, 0, stream>>>(
      (const __bf16*)hdnb, (const __bf16*)wt, (float*)d_out, cls_b2, Bg, 2752, 2752, 0);
}
